// RecurrentGCN_25735444038199
// MI455X (gfx1250) — matrix operand order repaired and checked by reading, not compiled, not run on hardware
//
#include <hip/hip_runtime.h>

typedef __attribute__((ext_vector_type(16))) _Float16 v16h;
typedef __attribute__((ext_vector_type(8)))  float    v8f;

#define N_NODES 200000
#define IN_DIM  16
#define HID_DIM 128
#define NTILES  (N_NODES / 16)   // 12500

#define LOG2E  1.4426950408889634f

__device__ __forceinline__ float fast_exp2(float x) { return __builtin_amdgcn_exp2f(x); }
__device__ __forceinline__ float fast_rcp(float x)  { return __builtin_amdgcn_rcpf(x); }

#if __has_builtin(__builtin_amdgcn_tanhf)
// Native gfx1250 v_tanh_f32: single TRANS32 op (confirmed in round-3 asm).
__device__ __forceinline__ float fast_tanh(float t) { return __builtin_amdgcn_tanhf(t); }
#else
// Fallback: tanh(t) = 1 - 2/(1 + exp(2t)), branch-free on raw trans ops.
__device__ __forceinline__ float fast_tanh(float t) {
    return __builtin_fmaf(-2.0f, fast_rcp(1.0f + fast_exp2(2.0f * LOG2E * t)), 1.0f);
}
#endif

// ---------------------------------------------------------------------------
// Prep: convert W_xz / W_xh (fp32 [16][128]) into WMMA B-fragment layout in
// f16, and fuse the bias pairs. The Z-gate weights/bias are pre-scaled by 0.5
// so the epilogue can use  1-sigmoid(s) = 0.5 - 0.5*tanh(s/2)  with a single
// v_tanh. B-matrix 32x16 f16 layout (wave32):
//   lanes 0-15 : half h (0..7) holds K=h, N = lane; h 8..15 -> K=16..23 zero
//   lanes16-31 : half h (0..7) holds K=8+h, N = lane-16; h 8..15 -> K=24..31 zero
// Fragment for column-tile t, lane L stored at frag[(t*32+L)*16 + h].
// ---------------------------------------------------------------------------
__global__ void __launch_bounds__(512) gru_prep_kernel(
    const float* __restrict__ W_xz, const float* __restrict__ b_xz,
    const float* __restrict__ b_hz,
    const float* __restrict__ W_xh, const float* __restrict__ b_xh,
    const float* __restrict__ b_hh,
    _Float16* __restrict__ fragZ, _Float16* __restrict__ fragH,
    float* __restrict__ bias_z, float* __restrict__ bias_h)
{
    const int tid = threadIdx.x;          // 0..511
    const int mat  = tid >> 8;            // 0 = Z (scaled 0.5), 1 = H
    const int rem  = tid & 255;
    const int t    = rem >> 5;            // column tile 0..7
    const int lane = rem & 31;

    const float   scale = mat ? 1.0f : 0.5f;
    const float*  W     = mat ? W_xh  : W_xz;
    _Float16*     frag  = (mat ? fragH : fragZ) + (t * 32 + lane) * 16;

    #pragma unroll
    for (int h = 0; h < 16; ++h) {
        float v = (h < 8) ? scale * W[((lane >> 4) * 8 + h) * HID_DIM + t * 16 + (lane & 15)] : 0.0f;
        frag[h] = (_Float16)v;
    }

    if (tid < HID_DIM) {
        bias_z[tid] = 0.5f * (b_xz[tid] + b_hz[tid]);   // pre-scaled for tanh form
        bias_h[tid] = b_xh[tid] + b_hh[tid];
    }
}

// ---------------------------------------------------------------------------
// Main: one wave per 16-node tile (grid-stride). Two v_wmma_f32_16x16x32_f16
// per column tile (update gate + candidate), v_tanh-only branch-free
// epilogue, [128]x[1] output projection folded into a shuffle reduction.
// Fully unrolled so every array is statically indexed and register-resident
// (no scratch spills, no v_movrels); >256-VGPR addressing via s_set_vgpr_msb
// is cheap SALU that co-issues.
// ---------------------------------------------------------------------------
__global__ void __launch_bounds__(256) gru_wmma_kernel(
    const float* __restrict__ x,
    const _Float16* __restrict__ fragZ,
    const _Float16* __restrict__ fragH,
    const float* __restrict__ bias_z,
    const float* __restrict__ bias_h,
    const float* __restrict__ W_out,
    const float* __restrict__ b_out,
    float* __restrict__ out)
{
    const int lane = threadIdx.x & 31;
    const int wave = threadIdx.x >> 5;
    const int waveGlobal = blockIdx.x * (blockDim.x >> 5) + wave;
    const int nWaves     = gridDim.x * (blockDim.x >> 5);

    // ---- constant B fragments + per-lane bias/Wout values (registers) ----
    v16h bZ[8], bH[8];
    #pragma unroll
    for (int t = 0; t < 8; ++t) {
        bZ[t] = *(const v16h*)(fragZ + (t * 32 + lane) * 16);
        bH[t] = *(const v16h*)(fragH + (t * 32 + lane) * 16);
    }
    const int nIdx = lane & 15;            // column within 16-wide tile
    float biasZ[8], biasH[8], wout[8];
    #pragma unroll
    for (int t = 0; t < 8; ++t) {
        biasZ[t] = bias_z[t * 16 + nIdx];
        biasH[t] = bias_h[t * 16 + nIdx];
        wout[t]  = W_out [t * 16 + nIdx];
    }
    const float bout = b_out[0];

    // A-fragment lane geometry: M = lane&15, K = koff + 0..7 (contiguous)
    const int M    = lane & 15;
    const int koff = (lane >> 4) * 8;

    // Upper K half (K=16..31) of the A fragment is loop-invariant zero.
    v16h a;
    #pragma unroll
    for (int h = 8; h < 16; ++h) a[h] = (_Float16)0.0f;

    for (int tile = waveGlobal; tile < NTILES; tile += nWaves) {
        const int nodeBase = tile * 16;

        // ---- A fragment: 16x16 of x, f32 -> f16 ----
        const float4* xrow = (const float4*)(x + (size_t)(nodeBase + M) * IN_DIM + koff);
        float4 x0 = xrow[0];
        float4 x1 = xrow[1];
        a[0] = (_Float16)x0.x; a[1] = (_Float16)x0.y;
        a[2] = (_Float16)x0.z; a[3] = (_Float16)x0.w;
        a[4] = (_Float16)x1.x; a[5] = (_Float16)x1.y;
        a[6] = (_Float16)x1.z; a[7] = (_Float16)x1.w;

        float partial[8];
        #pragma unroll
        for (int v = 0; v < 8; ++v) partial[v] = 0.0f;

        #pragma unroll
        for (int t = 0; t < 8; ++t) {
            v8f c = {};
            v8f accZ = __builtin_amdgcn_wmma_f32_16x16x32_f16(
                false, a, false, bZ[t], (short)0, c, false, false);
            v8f accH = __builtin_amdgcn_wmma_f32_16x16x32_f16(
                false, a, false, bH[t], (short)0, c, false, false);

            #pragma unroll
            for (int v = 0; v < 8; ++v) {
                // accZ computed at half scale: az = s/2
                float thz = fast_tanh(accZ[v] + biasZ[t]);
                float omz = __builtin_fmaf(-0.5f, thz, 0.5f);      // 1 - sigmoid(s)
                float ht  = fast_tanh(accH[v] + biasH[t]);         // candidate H~
                float hn  = fmaxf(omz * ht, 0.0f);                 // relu(H)
                partial[v] = __builtin_fmaf(hn, wout[t], partial[v]);
            }
        }

        // ---- reduce the 16 columns within each 16-lane half ----
        #pragma unroll
        for (int v = 0; v < 8; ++v) {
            float s = partial[v];
            s += __shfl_xor(s, 1, 32);
            s += __shfl_xor(s, 2, 32);
            s += __shfl_xor(s, 4, 32);
            s += __shfl_xor(s, 8, 32);
            partial[v] = s;
        }
        // lane 0 owns nodes base+0..7 (VGPR v -> M=v); lane 16 owns base+8..15
        if ((lane & 15) == 0) {
            const int mbase = nodeBase + ((lane >> 4) ? 8 : 0);
            #pragma unroll
            for (int v = 0; v < 8; ++v)
                out[mbase + v] = partial[v] + bout;
        }
    }
}

// ---------------------------------------------------------------------------
extern "C" void kernel_launch(void* const* d_in, const int* in_sizes, int n_in,
                              void* d_out, int out_size, void* d_ws, size_t ws_size,
                              hipStream_t stream) {
    // setup_inputs order:
    // 0:x 1:edge_index 2:edge_weight 3:W_xz 4:b_xz 5:W_hz 6:b_hz 7:W_xr 8:b_xr
    // 9:W_hr 10:b_hr 11:W_xh 12:b_xh 13:W_hh 14:b_hh 15:W_out 16:b_out
    const float* x     = (const float*)d_in[0];
    const float* W_xz  = (const float*)d_in[3];
    const float* b_xz  = (const float*)d_in[4];
    const float* b_hz  = (const float*)d_in[6];
    const float* W_xh  = (const float*)d_in[11];
    const float* b_xh  = (const float*)d_in[12];
    const float* b_hh  = (const float*)d_in[14];
    const float* W_out = (const float*)d_in[15];
    const float* b_out = (const float*)d_in[16];

    char* ws = (char*)d_ws;
    _Float16* fragZ = (_Float16*)(ws);             // 8 tiles * 32 lanes * 16 h * 2B = 8 KB
    _Float16* fragH = (_Float16*)(ws + 8192);      // 8 KB
    float* bias_z   = (float*)(ws + 16384);        // 512 B
    float* bias_h   = (float*)(ws + 16384 + 512);  // 512 B

    gru_prep_kernel<<<1, 512, 0, stream>>>(W_xz, b_xz, b_hz, W_xh, b_xh, b_hh,
                                           fragZ, fragH, bias_z, bias_h);

    // 256 blocks * 8 waves = 2048 waves; 12500 node-tiles -> ~6 tiles/wave,
    // amortizing the register-resident B fragments across the grid-stride loop.
    gru_wmma_kernel<<<256, 256, 0, stream>>>(x, fragZ, fragH, bias_z, bias_h,
                                             W_out, b_out, (float*)d_out);
}
